// PCVM_6511170421522
// MI455X (gfx1250) — compile-verified, matrix operand order checked by reading, not run on hardware
//
#include <hip/hip_runtime.h>
#include <hip/hip_bf16.h>

// ---------- types ----------
typedef __attribute__((ext_vector_type(16))) __bf16 v16bf;
typedef __attribute__((ext_vector_type(8)))  __bf16 v8bf;
typedef __attribute__((ext_vector_type(8)))  float  v8f;
typedef __attribute__((ext_vector_type(4)))  int    v4i_;

#define D_STATE 16
#define D_CONV 8
#define D_MODEL 128
#define D_INNER 256
#define DT_RANK 8
#define NSEQ 16              // 2 * batch(8) after channel split
#define LSEQ 4096            // 64*64
#define DBC_PAD 48           // 40 padded to 48 for 16-wide N tiles

// ---------- gfx1250 async global->LDS helpers (hedged) ----------
__device__ __forceinline__ void async_g2l_b128(const float* g, float* l) {
#if __has_builtin(__builtin_amdgcn_global_load_async_to_lds_b128)
  // param0: global (AS1) int4*  (non-const), param1: LDS (AS3) int4*.
  // Cast via integers: inttoptr is always legal, drops const, no addrspace-cast issues.
  __builtin_amdgcn_global_load_async_to_lds_b128(
      (__attribute__((address_space(1))) v4i_*)(unsigned long long)g,
      (__attribute__((address_space(3))) v4i_*)(unsigned int)(unsigned long long)l,
      0, 0);
#else
  l[0] = g[0]; l[1] = g[1]; l[2] = g[2]; l[3] = g[3];
#endif
}
__device__ __forceinline__ void wait_async_g2l() {
#if __has_builtin(__builtin_amdgcn_global_load_async_to_lds_b128)
#if __has_builtin(__builtin_amdgcn_s_wait_asynccnt)
  __builtin_amdgcn_s_wait_asynccnt(0);
#else
  asm volatile("s_wait_asynccnt 0" ::: "memory");
#endif
#endif
}

// ---------- WMMA tile core: one wave computes C[16 x 16*NT] ----------
template<int NT>
__device__ __forceinline__ void wmma_mm(const __bf16* __restrict__ Abase,
                                        const __bf16* __restrict__ Wbase,
                                        int K, v8f* acc) {
  const int lane = threadIdx.x & 31;
  const int mr   = lane & 15;      // row of A tile / col (n) of W tile
  const int half = lane >> 4;
  const __bf16* arow = Abase + (size_t)mr * K + half * 8;
  for (int k0 = 0; k0 < K; k0 += 32) {
    v16bf af;
    {
      v8bf lo = *(const v8bf*)(arow + k0);
      v8bf hi = *(const v8bf*)(arow + k0 + 16);
#pragma unroll
      for (int i = 0; i < 8; ++i) { af[i] = lo[i]; af[i + 8] = hi[i]; }
    }
#pragma unroll
    for (int t = 0; t < NT; ++t) {
      const __bf16* wrow = Wbase + (size_t)(t * 16 + mr) * K + half * 8;
      v16bf bfr;
      v8bf lo = *(const v8bf*)(wrow + k0);
      v8bf hi = *(const v8bf*)(wrow + k0 + 16);
#pragma unroll
      for (int i = 0; i < 8; ++i) { bfr[i] = lo[i]; bfr[i + 8] = hi[i]; }
      acc[t] = __builtin_amdgcn_wmma_f32_16x16x32_bf16(
          false, af, false, bfr, (short)0, acc[t], false, false);
    }
  }
}

// ---------- weight fp32 -> bf16 (with optional zero row padding) ----------
__global__ void cvt_w_kernel(const float* __restrict__ src, __bf16* __restrict__ dst,
                             int rows, int cols, int dst_rows) {
  int i = blockIdx.x * 256 + threadIdx.x;
  int total = dst_rows * cols;
  if (i >= total) return;
  int r = i / cols;
  dst[i] = (__bf16)(r < rows ? src[i] : 0.0f);
}

// ---------- LN1: x (8,256,4096) NCL -> xn (16,4096,128) bf16, channel-split ----------
__global__ void ln1_kernel(const float* __restrict__ x, const float* __restrict__ lnw,
                           const float* __restrict__ lnb, __bf16* __restrict__ xn) {
  __shared__ float tile[256][33];
  __shared__ float rs[32][8], rq[32][8];
  const int b  = blockIdx.x >> 7;          // 8 batches
  const int l0 = (blockIdx.x & 127) << 5;  // 128 chunks of 32 l
  const int tid = threadIdx.x;
  const float* xb = x + (size_t)b * 256 * LSEQ;
  for (int k = 0; k < 32; ++k) {
    int i = k * 256 + tid;
    int c = i >> 5, lo = i & 31;
    tile[c][lo] = xb[(size_t)c * LSEQ + l0 + lo];
  }
  __syncthreads();
  const int l = tid & 31, part = tid >> 5;   // 8 parts x 32 channels
  float s = 0.f, q = 0.f;
  for (int c = part * 32; c < part * 32 + 32; ++c) {
    float v = tile[c][l]; s += v; q += v * v;
  }
  rs[l][part] = s; rq[l][part] = q;
  __syncthreads();
  float sum = 0.f, sq = 0.f;
  for (int p = 0; p < 8; ++p) { sum += rs[l][p]; sq += rq[l][p]; }
  const float mean = sum * (1.0f / 256.0f);
  const float var  = sq * (1.0f / 256.0f) - mean * mean;
  const float inv  = rsqrtf(var + 1e-5f);
  for (int c = part * 32; c < part * 32 + 32; ++c) {
    float v = (tile[c][l] - mean) * inv * lnw[c] + lnb[c];
    int bb = (c < D_MODEL) ? b : b + 8;
    int d  = c & (D_MODEL - 1);
    xn[((size_t)bb * LSEQ + l0 + l) * D_MODEL + d] = (__bf16)v;
  }
}

// ---------- GEMM1: xz = xn @ in_proj_w^T ; split into xs_pre and silu(z) ----------
__global__ void gemm_inproj_kernel(const __bf16* __restrict__ xn, const __bf16* __restrict__ wA,
                                   __bf16* __restrict__ xs_pre, __bf16* __restrict__ zsil) {
  const int wave = (blockIdx.x * blockDim.x + threadIdx.x) >> 5;  // 32768 waves
  const int mt = wave & 4095;      // 4096 M tiles
  const int ng = wave >> 12;       // 8 groups of 64 cols
  v8f acc[4] = {};
  wmma_mm<4>(xn + (size_t)mt * 16 * D_MODEL, wA + (size_t)ng * 64 * D_MODEL, D_MODEL, acc);
  const int lane = threadIdx.x & 31;
  const int n0 = ng * 64 + (lane & 15);
  const int m0 = mt * 16 + (lane >> 4) * 8;
#pragma unroll
  for (int t = 0; t < 4; ++t) {
    const int e = n0 + t * 16;
#pragma unroll
    for (int r = 0; r < 8; ++r) {
      float v = acc[t][r];
      size_t row = (size_t)(m0 + r);
      if (e < D_INNER) {
        xs_pre[row * D_INNER + e] = (__bf16)v;
      } else {
        float sv = v / (1.0f + __expf(-v));   // silu(z)
        zsil[row * D_INNER + (e - D_INNER)] = (__bf16)sv;
      }
    }
  }
}

// ---------- causal depthwise conv(8) + bias + silu ----------
__global__ void conv_kernel(const __bf16* __restrict__ xs_pre, const float* __restrict__ cw,
                            const float* __restrict__ cb, __bf16* __restrict__ xs) {
  const int bb = blockIdx.x >> 4;
  const int l0 = (blockIdx.x & 15) << 8;    // 16 chunks of 256
  const int ch = threadIdx.x;
  float w[8];
#pragma unroll
  for (int j = 0; j < 8; ++j) w[j] = cw[ch * 8 + j];
  const float bias = cb[ch];
  float win[8];
#pragma unroll
  for (int j = 0; j < 7; ++j) {
    int l = l0 - 7 + j;
    win[j] = (l >= 0) ? (float)xs_pre[((size_t)bb * LSEQ + l) * D_INNER + ch] : 0.0f;
  }
  for (int li = 0; li < 256; ++li) {
    const int l = l0 + li;
    win[7] = (float)xs_pre[((size_t)bb * LSEQ + l) * D_INNER + ch];
    float a = bias;
#pragma unroll
    for (int j = 0; j < 8; ++j) a += w[j] * win[j];
    float s = a / (1.0f + __expf(-a));       // silu
    xs[((size_t)bb * LSEQ + l) * D_INNER + ch] = (__bf16)s;
#pragma unroll
    for (int j = 0; j < 7; ++j) win[j] = win[j + 1];
  }
}

// ---------- GEMM2: dbc = xs @ x_proj_w^T (N padded to 48) ----------
__global__ void gemm_xproj_kernel(const __bf16* __restrict__ xs, const __bf16* __restrict__ wX,
                                  float* __restrict__ dbc) {
  const int wave = (blockIdx.x * blockDim.x + threadIdx.x) >> 5;  // 4096 waves
  const int mt = wave;
  v8f acc[3] = {};
  wmma_mm<3>(xs + (size_t)mt * 16 * D_INNER, wX, D_INNER, acc);
  const int lane = threadIdx.x & 31;
  const int n0 = lane & 15;
  const int m0 = mt * 16 + (lane >> 4) * 8;
#pragma unroll
  for (int t = 0; t < 3; ++t)
#pragma unroll
    for (int r = 0; r < 8; ++r)
      dbc[(size_t)(m0 + r) * DBC_PAD + t * 16 + n0] = acc[t][r];
}

// ---------- selective scan, state-parallel ----------
// 16 lanes per (bb,ch): lane owns state n. block = 256 thr = 16 channels x 16 states.
// grid = NSEQ * 16 blocks. Recurrence kept in one register per lane; y via shfl-xor
// reduction over the 16-lane state group (xor masks 1/2/4/8 stay inside the group).
#define SCHUNK 64
__global__ void scan_kernel(const float* __restrict__ dbc, const __bf16* __restrict__ xs,
                            const __bf16* __restrict__ zsil, const float* __restrict__ wdt_g,
                            const float* __restrict__ dtb_g, const float* __restrict__ alog,
                            const float* __restrict__ Dv, __bf16* __restrict__ y) {
  __shared__ __align__(16) float sdbc[SCHUNK][DBC_PAD];
  const int bb = blockIdx.x >> 4;                 // 16 sequences
  const int chbase = (blockIdx.x & 15) << 4;      // 16 channel groups
  const int tid = threadIdx.x;
  const int n  = tid & 15;                        // state index
  const int ch = chbase + (tid >> 4);             // channel
  float wdt[DT_RANK];
#pragma unroll
  for (int r = 0; r < DT_RANK; ++r) wdt[r] = wdt_g[ch * DT_RANK + r];
  const float dtb = dtb_g[ch];
  const float an  = -__expf(alog[ch * D_STATE + n]);
  const float Dch = Dv[ch];
  float h = 0.f;
  const float* dbcb = dbc + (size_t)bb * LSEQ * DBC_PAD;
  for (int t0 = 0; t0 < LSEQ; t0 += SCHUNK) {
    __syncthreads();
    // stage 64 timesteps of (dt[8] | B[16] | C[16] | pad) : 3072 floats, 16B per op
#pragma unroll
    for (int k = 0; k < (SCHUNK * DBC_PAD) / (256 * 4); ++k) {
      int i = (k * 256 + tid) * 4;
      async_g2l_b128(dbcb + (size_t)t0 * DBC_PAD + i, &sdbc[0][0] + i);
    }
    wait_async_g2l();
    __syncthreads();
    if (t0 + SCHUNK < LSEQ)   // prefetch next chunk into cache
      __builtin_prefetch(dbcb + (size_t)(t0 + SCHUNK) * DBC_PAD + ((tid & 15) << 6), 0, 1);
    for (int tt = 0; tt < SCHUNK; ++tt) {
      const int t = t0 + tt;
      // softplus(dt . Wdt + b), redundantly on the 16 lanes of this channel
      float d = dtb;
#pragma unroll
      for (int r = 0; r < DT_RANK; ++r) d += sdbc[tt][r] * wdt[r];
      d = (d > 20.f) ? d : __logf(1.0f + __expf(d));
      const size_t idx = ((size_t)bb * LSEQ + t) * D_INNER + ch;
      const float xv = (float)xs[idx];
      h = h * __expf(d * an) + (d * xv) * sdbc[tt][DT_RANK + n];
      float pv = h * sdbc[tt][DT_RANK + D_STATE + n];
      pv += __shfl_xor(pv, 1, 32);
      pv += __shfl_xor(pv, 2, 32);
      pv += __shfl_xor(pv, 4, 32);
      pv += __shfl_xor(pv, 8, 32);
      if (n == 0) {
        float yv = (pv + Dch * xv) * (float)zsil[idx];
        y[idx] = (__bf16)yv;
      }
    }
  }
}

// ---------- GEMM3: ym = y @ out_proj_w^T, scattered into ycat (8,4096,256) ----------
__global__ void gemm_outproj_kernel(const __bf16* __restrict__ y, const __bf16* __restrict__ wO,
                                    float* __restrict__ ycat) {
  const int wave = (blockIdx.x * blockDim.x + threadIdx.x) >> 5;  // 8192 waves
  const int mt = wave & 4095;
  const int ng = wave >> 12;       // 2 groups of 64 cols
  v8f acc[4] = {};
  wmma_mm<4>(y + (size_t)mt * 16 * D_INNER, wO + (size_t)ng * 64 * D_INNER, D_INNER, acc);
  const int lane = threadIdx.x & 31;
  const int n0 = ng * 64 + (lane & 15);
  const int m0 = mt * 16 + (lane >> 4) * 8;
#pragma unroll
  for (int t = 0; t < 4; ++t)
#pragma unroll
    for (int r = 0; r < 8; ++r) {
      const int row = m0 + r;
      const int bbv = row >> 12, l = row & (LSEQ - 1);
      const int b = bbv & 7, cbase = (bbv >> 3) * D_MODEL;
      ycat[((size_t)b * LSEQ + l) * 256 + cbase + n0 + t * 16] = acc[t][r];
    }
}

// ---------- LN2 over (8*4096, 256) ----------
__global__ void ln2_kernel(const float* __restrict__ yc, const float* __restrict__ lnw,
                           const float* __restrict__ lnb, __bf16* __restrict__ out) {
  __shared__ float s1[256], s2[256];
  const int row = blockIdx.x;
  const int c = threadIdx.x;
  const float v = yc[(size_t)row * 256 + c];
  s1[c] = v; s2[c] = v * v;
  __syncthreads();
  for (int off = 128; off > 0; off >>= 1) {
    if (c < off) { s1[c] += s1[c + off]; s2[c] += s2[c + off]; }
    __syncthreads();
  }
  const float mean = s1[0] * (1.0f / 256.0f);
  const float var  = s2[0] * (1.0f / 256.0f) - mean * mean;
  const float inv  = rsqrtf(var + 1e-5f);
  out[(size_t)row * 256 + c] = (__bf16)((v - mean) * inv * lnw[c] + lnb[c]);
}

// ---------- GEMM4: out = ycn @ proj_w^T + b, transposed store to NCHW ----------
__global__ void gemm_final_kernel(const __bf16* __restrict__ ycn, const __bf16* __restrict__ wP,
                                  const float* __restrict__ pb, float* __restrict__ out) {
  const int wave = (blockIdx.x * blockDim.x + threadIdx.x) >> 5;  // 8192 waves
  const int mt = wave & 2047;       // 2048 M tiles (M = 32768)
  const int ng = wave >> 11;        // 4 groups of 64 cols
  v8f acc[4] = {};
  wmma_mm<4>(ycn + (size_t)mt * 16 * 256, wP + (size_t)ng * 64 * 256, 256, acc);
  const int lane = threadIdx.x & 31;
  const int n0 = ng * 64 + (lane & 15);
  const int m0 = mt * 16 + (lane >> 4) * 8;
#pragma unroll
  for (int t = 0; t < 4; ++t) {
    const int o = n0 + t * 16;
    const float bias = pb[o];
#pragma unroll
    for (int r = 0; r < 8; ++r) {
      const int row = m0 + r;
      const int b = row >> 12, l = row & (LSEQ - 1);
      out[((size_t)b * 256 + o) * LSEQ + l] = acc[t][r] + bias;
    }
  }
}

// ---------- host ----------
static inline char* carve(char*& p, size_t bytes) {
  char* r = p;
  p += (bytes + 255) & ~(size_t)255;
  return r;
}

extern "C" void kernel_launch(void* const* d_in, const int* in_sizes, int n_in,
                              void* d_out, int out_size, void* d_ws, size_t ws_size,
                              hipStream_t stream) {
  const float* x        = (const float*)d_in[0];
  const float* ln_w     = (const float*)d_in[1];
  const float* ln_b     = (const float*)d_in[2];
  const float* in_projw = (const float*)d_in[3];
  const float* conv_w   = (const float*)d_in[4];
  const float* conv_b   = (const float*)d_in[5];
  const float* x_projw  = (const float*)d_in[6];
  const float* dt_projw = (const float*)d_in[7];
  const float* dt_projb = (const float*)d_in[8];
  const float* A_log    = (const float*)d_in[9];
  const float* Dvec     = (const float*)d_in[10];
  const float* out_projw= (const float*)d_in[11];
  const float* proj_w   = (const float*)d_in[12];
  const float* proj_b   = (const float*)d_in[13];
  float* out = (float*)d_out;

  char* p = (char*)d_ws;
  __bf16* xn     = (__bf16*)carve(p, (size_t)NSEQ * LSEQ * D_MODEL * 2);
  __bf16* wA     = (__bf16*)carve(p, (size_t)512 * 128 * 2);
  __bf16* wX     = (__bf16*)carve(p, (size_t)DBC_PAD * 256 * 2);
  __bf16* wO     = (__bf16*)carve(p, (size_t)128 * 256 * 2);
  __bf16* wP     = (__bf16*)carve(p, (size_t)256 * 256 * 2);
  __bf16* xs_pre = (__bf16*)carve(p, (size_t)NSEQ * LSEQ * D_INNER * 2);
  __bf16* zsil   = (__bf16*)carve(p, (size_t)NSEQ * LSEQ * D_INNER * 2);
  __bf16* xs     = (__bf16*)carve(p, (size_t)NSEQ * LSEQ * D_INNER * 2);
  float*  dbc    = (float*) carve(p, (size_t)NSEQ * LSEQ * DBC_PAD * 4);
  __bf16* ybuf   = (__bf16*)carve(p, (size_t)NSEQ * LSEQ * D_INNER * 2);
  float*  ycat   = (float*) carve(p, (size_t)8 * LSEQ * 256 * 4);
  __bf16* ycn    = (__bf16*)carve(p, (size_t)8 * LSEQ * 256 * 2);

  // weight conversions (bf16, x_proj zero-padded 40 -> 48 rows)
  cvt_w_kernel<<<(512 * 128 + 255) / 256, 256, 0, stream>>>(in_projw, wA, 512, 128, 512);
  cvt_w_kernel<<<(DBC_PAD * 256 + 255) / 256, 256, 0, stream>>>(x_projw, wX, 40, 256, DBC_PAD);
  cvt_w_kernel<<<(128 * 256 + 255) / 256, 256, 0, stream>>>(out_projw, wO, 128, 256, 128);
  cvt_w_kernel<<<(256 * 256 + 255) / 256, 256, 0, stream>>>(proj_w, wP, 256, 256, 256);

  ln1_kernel<<<8 * 128, 256, 0, stream>>>(x, ln_w, ln_b, xn);
  gemm_inproj_kernel<<<4096, 256, 0, stream>>>(xn, wA, xs_pre, zsil);
  conv_kernel<<<NSEQ * 16, 256, 0, stream>>>(xs_pre, conv_w, conv_b, xs);
  gemm_xproj_kernel<<<512, 256, 0, stream>>>(xs, wX, dbc);
  scan_kernel<<<NSEQ * 16, 256, 0, stream>>>(dbc, xs, zsil, dt_projw, dt_projb, A_log, Dvec, ybuf);
  gemm_outproj_kernel<<<1024, 256, 0, stream>>>(ybuf, wO, ycat);
  ln2_kernel<<<8 * LSEQ, 256, 0, stream>>>(ycat, ln_w, ln_b, ycn);
  gemm_final_kernel<<<1024, 256, 0, stream>>>(ycn, wP, proj_b, out);
}
